// VQVAE_10746008174971
// MI455X (gfx1250) — matrix operand order repaired and checked by reading, not compiled, not run on hardware
//
#include <hip/hip_runtime.h>
#include <hip/hip_bf16.h>
#include <math.h>

// ---------------------------------------------------------------------------
// CDNA5 (gfx1250) wave32 WMMA types & helpers
// ---------------------------------------------------------------------------
typedef __attribute__((ext_vector_type(16))) _Float16 v16h;
typedef __attribute__((ext_vector_type(8)))  _Float16 v8h;
typedef __attribute__((ext_vector_type(8)))  float    v8f;
typedef int v4i __attribute__((vector_size(16)));

#ifndef __has_builtin
#define __has_builtin(x) 0
#endif
#if __has_builtin(__builtin_amdgcn_global_load_async_to_lds_b128) && \
    __has_builtin(__builtin_amdgcn_s_wait_asynccnt)
#define USE_ASYNC_LDS 1
#else
#define USE_ASYNC_LDS 0
#endif

__device__ __forceinline__ v8f wmma_f16(v16h a, v16h b, v8f c) {
  // D = A(16x32,f16) * B(32x16,f16) + C(16x16,f32)
  return __builtin_amdgcn_wmma_f32_16x16x32_f16(
      /*neg_a=*/false, a, /*neg_b=*/false, b,
      /*c_mod=*/(short)0, c, /*reuse_a=*/false, /*reuse_b=*/false);
}

__device__ __forceinline__ v16h cat8(v8h lo, v8h hi) {
  return __builtin_shufflevector(lo, hi, 0,1,2,3,4,5,6,7,8,9,10,11,12,13,14,15);
}

// ---------------------------------------------------------------------------
// Weight conversion / packing kernels
// ---------------------------------------------------------------------------
__global__ void cvt_f16_kern(const float* __restrict__ s, _Float16* __restrict__ d, int n) {
  int i = blockIdx.x * blockDim.x + threadIdx.x;
  if (i < n) d[i] = (_Float16)s[i];
}

// torch ConvTranspose2d weight (IC, OC, 4, 4) -> per output parity class
// A matrix [par][OC][IC*4] with taps (j selects ky, i selects kx)
__global__ void pack_deconv_kern(const float* __restrict__ w, _Float16* __restrict__ wpk,
                                 int IC, int OC) {
  int idx = blockIdx.x * blockDim.x + threadIdx.x;
  int total = 4 * OC * IC * 4;
  if (idx >= total) return;
  int tmp = idx;
  int t  = tmp & 3; tmp >>= 2;
  int ic = tmp % IC; tmp /= IC;
  int oc = tmp % OC;
  int par = tmp / OC;
  int py = par >> 1, px = par & 1;
  int j = t >> 1, i = t & 1;
  // py=0: ky in {1,3} (dy {0,-1}); py=1: ky in {0,2} (dy {+1,0})
  int ky = py ? (j ? 2 : 0) : (j ? 3 : 1);
  int kx = px ? (i ? 2 : 0) : (i ? 3 : 1);
  wpk[idx] = (_Float16)w[((size_t)(ic * OC + oc) * 4 + ky) * 4 + kx];
}

__global__ void cbnorm_kern(const float* __restrict__ cb, float* __restrict__ cbn) {
  int i = blockIdx.x * blockDim.x + threadIdx.x;
  if (i < 1024) {
    const float* r = cb + (size_t)i * 64;
    float s = 0.f;
    for (int c = 0; c < 64; ++c) s += r[c] * r[c];
    cbn[i] = s;
  }
}

// ---------------------------------------------------------------------------
// Implicit-GEMM conv (NCHW), M=OC, N=pixels, K=IC*KH*KW (K % 64 == 0).
// Each wave: 16(M) x 32(N) tile (one A frag -> two B frags / two accs).
// 4 waves/WG -> 128 pixels per workgroup. All spatial sizes are powers of 2.
// ---------------------------------------------------------------------------
template <int KH, int KW, int STRIDE, int PAD>
__global__ void __launch_bounds__(128)
conv_wmma_kern(const float* __restrict__ in, const _Float16* __restrict__ wt,
               const float* __restrict__ bias, float* __restrict__ out,
               int IC, int IH, int lgIW, int OC, int lgOW, int HWo) {
  constexpr int KHW = KH * KW;
  const int lane = threadIdx.x & 31;
  const int wid  = threadIdx.x >> 5;
  const int lh   = lane >> 4;      // lane half
  const int lm   = lane & 15;
  const int n    = blockIdx.z;
  const int oc0  = blockIdx.y << 4;
  const int IW   = 1 << lgIW;
  const int OW   = 1 << lgOW;
  const int p0   = (blockIdx.x << 7) + (wid << 5);   // wave's 32 pixels
  const int pA   = p0 + lm;
  const int pB   = p0 + 16 + lm;
  const int K    = IC * KHW;
  const int HWin = IH << lgIW;
  const float* __restrict__ inN = in + (size_t)n * IC * HWin;
  const _Float16* __restrict__ arow = wt + (size_t)(oc0 + lm) * K;
  const int iyA = (pA >> lgOW) * STRIDE - PAD, ixA = (pA & (OW - 1)) * STRIDE - PAD;
  const int iyB = (pB >> lgOW) * STRIDE - PAD, ixB = (pB & (OW - 1)) * STRIDE - PAD;

  auto buildB = [&](int iy0, int ix0, int k0) -> v16h {
    v16h b;
#pragma unroll
    for (int e = 0; e < 16; ++e) {
      int k  = k0 + e + (e & 8) + (lh << 3);  // table-order K per lane-half
      int ic = k / KHW;                 // constexpr KHW -> shift
      int r  = k - ic * KHW;
      int ky = r / KW;                  // constexpr KW -> shift
      int kx = r - ky * KW;
      int iy = iy0 + ky;
      int ix = ix0 + kx;
      float v = 0.f;
      if ((unsigned)iy < (unsigned)IH && (unsigned)ix < (unsigned)IW)
        v = inN[(size_t)ic * HWin + ((size_t)iy << lgIW) + ix];
      b[e] = (_Float16)v;
    }
    return b;
  };

  v8f acc0 = {0.f, 0.f, 0.f, 0.f, 0.f, 0.f, 0.f, 0.f};
  v8f acc1 = {0.f, 0.f, 0.f, 0.f, 0.f, 0.f, 0.f, 0.f};
#pragma unroll 2
  for (int k0 = 0; k0 < K; k0 += 32) {
    __builtin_prefetch(arow + k0 + 128, 0, 1);   // global_prefetch_b8 on weight stream
    // A fragment: lane holds row (oc0+lm); two contiguous 8-elem K runs
    v16h a = cat8(*(const v8h*)(arow + k0 + lh * 8),
                  *(const v8h*)(arow + k0 + 16 + lh * 8));
    acc0 = wmma_f16(a, buildB(iyA, ixA, k0), acc0);
    acc1 = wmma_f16(a, buildB(iyB, ixB, k0), acc1);
  }
  // D layout: VGPR r -> M = r + lh*8, lane lm -> N = pixel
  float* __restrict__ outN = out + (size_t)n * OC * HWo;
#pragma unroll
  for (int r = 0; r < 8; ++r) {
    int oc = oc0 + r + (lh << 3);
    outN[(size_t)oc * HWo + pA] = acc0[r] + bias[oc];
    outN[(size_t)oc * HWo + pB] = acc1[r] + bias[oc];
  }
}

// ---------------------------------------------------------------------------
// 1x1 conv at 32x32 (HW=1024): GEMM with contiguous B rows.
// B tiles (32K x 128N fp32, 16KB) are staged into LDS with async
// global->LDS copies (ASYNCcnt) when the toolchain exposes them, double
// buffered across the K loop.  A (f16 weights) loads stay direct.
// ---------------------------------------------------------------------------
__global__ void __launch_bounds__(128)
conv1x1_wmma_kern(const float* __restrict__ in, const _Float16* __restrict__ wt,
                  const float* __restrict__ bias, float* __restrict__ out,
                  int IC, int OC) {
  __shared__ float btile[2][32 * 128];
  const int tid  = threadIdx.x;
  const int lane = tid & 31;
  const int wid  = tid >> 5;
  const int lh   = lane >> 4;
  const int lm   = lane & 15;
  const int n    = blockIdx.z;
  const int oc0  = blockIdx.y << 4;
  const int p0   = blockIdx.x << 7;       // workgroup pixel base
  const int pw   = wid << 5;              // wave pixel offset inside tile
  const float* __restrict__ inN = in + ((size_t)n * IC << 10);
  const _Float16* __restrict__ arow = wt + (size_t)(oc0 + lm) * IC;

  auto stage = [&](int k0, int buf) {
    // 128 threads x 8 iters x 16B = 16KB tile [32 K rows][128 pixels]
#pragma unroll
    for (int it = 0; it < 8; ++it) {
      int elem = (it * 128 + tid) << 2;   // float index within tile
      int row  = elem >> 7;
      int col  = elem & 127;
      const float* g = inN + (((size_t)(k0 + row)) << 10) + p0 + col;
#if USE_ASYNC_LDS
      float* gm = (float*)g;              // drop const for the builtin
      __builtin_amdgcn_global_load_async_to_lds_b128(
          (__attribute__((address_space(1))) v4i*)gm,
          (__attribute__((address_space(3))) v4i*)(&btile[buf][elem]), 0, 0);
#else
      *(float4*)(&btile[buf][elem]) = *(const float4*)g;
#endif
    }
  };

  v8f acc0 = {0.f, 0.f, 0.f, 0.f, 0.f, 0.f, 0.f, 0.f};
  v8f acc1 = {0.f, 0.f, 0.f, 0.f, 0.f, 0.f, 0.f, 0.f};
  const int nk = IC >> 5;
  stage(0, 0);
  for (int t = 0; t < nk; ++t) {
    const int k0  = t << 5;
    const int buf = t & 1;
    if (t + 1 < nk) {
      stage(k0 + 32, buf ^ 1);
#if USE_ASYNC_LDS
      __builtin_amdgcn_s_wait_asynccnt(8);   // drain previous tile's 8 issues
#endif
    } else {
#if USE_ASYNC_LDS
      __builtin_amdgcn_s_wait_asynccnt(0);
#endif
    }
    __syncthreads();                         // tile visible to all waves
    v16h a = cat8(*(const v8h*)(arow + k0 + lh * 8),
                  *(const v8h*)(arow + k0 + 16 + lh * 8));
    const float* bt = btile[buf];
    v16h bA, bB;
#pragma unroll
    for (int e = 0; e < 16; ++e) {
      int k = e + (e & 8) + (lh << 3);       // B-fragment K map
      bA[e] = (_Float16)bt[(k << 7) + pw + lm];
      bB[e] = (_Float16)bt[(k << 7) + pw + 16 + lm];
    }
    acc0 = wmma_f16(a, bA, acc0);
    acc1 = wmma_f16(a, bB, acc1);
    __syncthreads();                         // protect buf before next overwrite
  }
  float* __restrict__ outN = out + ((size_t)n * OC << 10);
  const int pA = p0 + pw + lm;
  const int pB = pA + 16;
#pragma unroll
  for (int r = 0; r < 8; ++r) {
    int oc = oc0 + r + (lh << 3);
    outN[((size_t)oc << 10) + pA] = acc0[r] + bias[oc];
    outN[((size_t)oc << 10) + pB] = acc1[r] + bias[oc];
  }
}

// ---------------------------------------------------------------------------
// ConvTranspose2d k4 s2 p1 as 4 parity-class implicit GEMMs, K = IC*4
// grid.z = batch*4 + parity; N = input-resolution pixels of this parity class
// ---------------------------------------------------------------------------
__global__ void __launch_bounds__(128)
deconv_wmma_kern(const float* __restrict__ in, const _Float16* __restrict__ wpk,
                 const float* __restrict__ bias, float* __restrict__ out,
                 int IC, int IH, int lgIW, int OC) {
  const int lane = threadIdx.x & 31;
  const int wid  = threadIdx.x >> 5;
  const int lh   = lane >> 4;
  const int lm   = lane & 15;
  const int zz   = blockIdx.z;
  const int n    = zz >> 2;
  const int par  = zz & 3;
  const int py = par >> 1, px = par & 1;
  const int oc0  = blockIdx.y << 4;
  const int IW   = 1 << lgIW;
  const int p0   = (blockIdx.x << 7) + (wid << 5);
  const int pA   = p0 + lm;
  const int pB   = p0 + 16 + lm;
  const int ayA = pA >> lgIW, bxA = pA & (IW - 1);
  const int ayB = pB >> lgIW, bxB = pB & (IW - 1);
  const int K = IC << 2;
  const int HWin = IH << lgIW;
  const int OH = IH << 1;
  const float* __restrict__ inN = in + (size_t)n * IC * HWin;
  const _Float16* __restrict__ arow = wpk + ((size_t)par * OC + (oc0 + lm)) * K;
  const int dyA = py ? 1 : 0, dyB = py ? 0 : -1;
  const int dxA = px ? 1 : 0, dxB = px ? 0 : -1;

  auto buildB = [&](int ay, int bx, int k0) -> v16h {
    v16h b;
#pragma unroll
    for (int e = 0; e < 16; ++e) {
      int k  = k0 + e + (e & 8) + (lh << 3);
      int ic = k >> 2;
      int t  = k & 3;
      int dy = (t >> 1) ? dyB : dyA;
      int dx = (t & 1)  ? dxB : dxA;
      int iy = ay + dy, ix = bx + dx;
      float v = 0.f;
      if ((unsigned)iy < (unsigned)IH && (unsigned)ix < (unsigned)IW)
        v = inN[(size_t)ic * HWin + ((size_t)iy << lgIW) + ix];
      b[e] = (_Float16)v;
    }
    return b;
  };

  v8f acc0 = {0.f, 0.f, 0.f, 0.f, 0.f, 0.f, 0.f, 0.f};
  v8f acc1 = {0.f, 0.f, 0.f, 0.f, 0.f, 0.f, 0.f, 0.f};
#pragma unroll 2
  for (int k0 = 0; k0 < K; k0 += 32) {
    __builtin_prefetch(arow + k0 + 128, 0, 1);
    v16h a = cat8(*(const v8h*)(arow + k0 + lh * 8),
                  *(const v8h*)(arow + k0 + 16 + lh * 8));
    acc0 = wmma_f16(a, buildB(ayA, bxA, k0), acc0);
    acc1 = wmma_f16(a, buildB(ayB, bxB, k0), acc1);
  }
  float* __restrict__ outN = out + ((size_t)n * OC * HWin << 2);
  const int oyA = (ayA << 1) + py, oxA = (bxA << 1) + px;
  const int oyB = (ayB << 1) + py, oxB = (bxB << 1) + px;
#pragma unroll
  for (int r = 0; r < 8; ++r) {
    int oc = oc0 + r + (lh << 3);
    float bi = bias[oc];
    outN[(((size_t)oc * OH + oyA) << (lgIW + 1)) + oxA] = acc0[r] + bi;
    outN[(((size_t)oc * OH + oyB) << (lgIW + 1)) + oxB] = acc1[r] + bi;
  }
}

// ---------------------------------------------------------------------------
// Stem 1x1 conv 3->64 (K too small for WMMA; bandwidth bound). HW = 65536.
// ---------------------------------------------------------------------------
__global__ void stem_kern(const float* __restrict__ x, const float* __restrict__ w,
                          const float* __restrict__ b, float* __restrict__ o) {
  size_t i = (size_t)blockIdx.x * blockDim.x + threadIdx.x;
  int p  = (int)(i & 65535);
  size_t t = i >> 16;
  int oc = (int)(t & 63);
  int n  = (int)(t >> 6);
  const float* xn = x + ((size_t)n * 3 << 16) + p;
  o[i] = b[oc] + w[oc * 3 + 0] * xn[0] + w[oc * 3 + 1] * xn[65536] + w[oc * 3 + 2] * xn[131072];
}

// ---------------------------------------------------------------------------
// Training-mode BN: per-channel biased stats -> scale/shift, then in-place SiLU
// ---------------------------------------------------------------------------
__global__ void bn_stats_kern(const float* __restrict__ x, const float* __restrict__ g,
                              const float* __restrict__ be, float* __restrict__ sc,
                              int C, int lgHW) {
  int c = blockIdx.x;
  int HW = 1 << lgHW;
  int tot = 8 << lgHW;
  float s = 0.f, s2 = 0.f;
  for (int i = threadIdx.x; i < tot; i += blockDim.x) {
    int n = i >> lgHW, j = i & (HW - 1);
    float v = x[((size_t)(n * C + c) << lgHW) + j];
    s += v; s2 += v * v;
  }
  __shared__ float rs[256], rq[256];
  rs[threadIdx.x] = s; rq[threadIdx.x] = s2;
  __syncthreads();
  for (int k = 128; k > 0; k >>= 1) {
    if ((int)threadIdx.x < k) { rs[threadIdx.x] += rs[threadIdx.x + k]; rq[threadIdx.x] += rq[threadIdx.x + k]; }
    __syncthreads();
  }
  if (threadIdx.x == 0) {
    float inv = 1.f / (float)tot;
    float m = rs[0] * inv;
    float var = rq[0] * inv - m * m;
    float scale = g[c] * rsqrtf(var + 1e-5f);
    sc[2 * c]     = scale;
    sc[2 * c + 1] = be[c] - m * scale;
  }
}

__global__ void bn_silu_kern(float* __restrict__ x, const float* __restrict__ sc,
                             int C, int lgHW, long long total) {
  long long i = (long long)blockIdx.x * blockDim.x + threadIdx.x;
  if (i >= total) return;
  int c = (int)((i >> lgHW) & (C - 1));     // C is a power of two
  float y = x[i] * sc[2 * c] + sc[2 * c + 1];
  x[i] = y / (1.f + expf(-y));
}

// ---------------------------------------------------------------------------
// VQ: argmin_k ( |c_k|^2 - 2 z.c_k ) via WMMA. z: [8][64][1024] NCHW-flat.
// One wave: 16 pixels x all 1024 codes; A frags cached across code tiles.
// ---------------------------------------------------------------------------
__global__ void __launch_bounds__(128)
vq_argmin_kern(const float* __restrict__ zf, const _Float16* __restrict__ cbh,
               const float* __restrict__ cbn, int* __restrict__ idxb) {
  const int lane = threadIdx.x & 31;
  const int wid  = threadIdx.x >> 5;
  const int lh = lane >> 4, lm = lane & 15;
  const int n  = blockIdx.z;
  const int p0 = (blockIdx.x << 6) + (wid << 4);
  const float* __restrict__ zn = zf + ((size_t)n << 16);

  v16h a0, a1;
#pragma unroll
  for (int e = 0; e < 16; ++e) {
    int c = e + lh * 16;                          // A-fragment K map
    a0[e] = (_Float16)zn[((size_t)c << 10) + p0 + lm];
    a1[e] = (_Float16)zn[((size_t)(c + 32) << 10) + p0 + lm];
  }
  float best[8]; int bidx[8];
#pragma unroll
  for (int r = 0; r < 8; ++r) { best[r] = 3.4e38f; bidx[r] = 0; }

#pragma unroll 2
  for (int nt = 0; nt < 64; ++nt) {
    const int c0 = nt << 4;
    const _Float16* brow = cbh + ((size_t)(c0 + lm) << 6);  // code row for this lane's column
    v16h b0 = cat8(*(const v8h*)(brow + lh * 16),      *(const v8h*)(brow + lh * 16 + 8));
    v16h b1 = cat8(*(const v8h*)(brow + 32 + lh * 16), *(const v8h*)(brow + 32 + lh * 16 + 8));
    v8f acc = {0.f, 0.f, 0.f, 0.f, 0.f, 0.f, 0.f, 0.f};
    acc = wmma_f16(a0, b0, acc);
    acc = wmma_f16(a1, b1, acc);
    float cn = cbn[c0 + lm];
#pragma unroll
    for (int r = 0; r < 8; ++r) {
      float s = cn - 2.f * acc[r];
      if (s < best[r]) { best[r] = s; bidx[r] = c0 + lm; }
    }
  }
  // wave32 cross-lane argmin within each lane-half (columns of a row)
#pragma unroll
  for (int r = 0; r < 8; ++r) {
#pragma unroll
    for (int m = 1; m < 16; m <<= 1) {
      float ov = __shfl_xor(best[r], m, 32);
      int   oi = __shfl_xor(bidx[r], m, 32);
      if (ov < best[r] || (ov == best[r] && oi < bidx[r])) { best[r] = ov; bidx[r] = oi; }
    }
  }
  if (lm == 0) {
#pragma unroll
    for (int r = 0; r < 8; ++r)
      idxb[(n << 10) + p0 + r + (lh << 3)] = bidx[r];
  }
}

__global__ void zero_kern(float* p) { if (threadIdx.x == 0 && blockIdx.x == 0) *p = 0.f; }

// Gather q (NCHW) + accumulate sum((q - z)^2).  Forward value of
// codebook_loss == commitment_loss, so total loss = (1+BETA)*MSE.
__global__ void vq_gather_kern(const float* __restrict__ z, const float* __restrict__ cb,
                               const int* __restrict__ idx, float* __restrict__ q,
                               float* __restrict__ lossacc) {
  int t = blockIdx.x * blockDim.x + threadIdx.x;
  float part = 0.f;
  if (t < 8192) {
    int n = t >> 10, p = t & 1023;
    int id = idx[t];
    const float* crow = cb + ((size_t)id << 6);
    for (int c = 0; c < 64; ++c) {
      float cv = crow[c];
      size_t zi = ((((size_t)n << 6) + c) << 10) + p;
      float zv = z[zi];
      q[zi] = cv;
      float d = cv - zv;
      part += d * d;
    }
  }
  __shared__ float red[256];
  red[threadIdx.x] = part;
  __syncthreads();
  for (int s = 128; s > 0; s >>= 1) {
    if ((int)threadIdx.x < s) red[threadIdx.x] += red[threadIdx.x + s];
    __syncthreads();
  }
  if (threadIdx.x == 0) atomicAdd(lossacc, red[0]);
}

__global__ void loss_fin_kern(const float* __restrict__ acc, float* __restrict__ out) {
  if (threadIdx.x == 0 && blockIdx.x == 0)
    out[0] = acc[0] * (1.0f + 0.2f) / (8.0f * 1024.0f * 64.0f);
}

// ---------------------------------------------------------------------------
// Head: 1x1 conv 64->3 + tanh (OC too small for WMMA; bandwidth bound)
// ---------------------------------------------------------------------------
__global__ void head_kern(const float* __restrict__ d, const float* __restrict__ w,
                          const float* __restrict__ b, float* __restrict__ o) {
  int t = blockIdx.x * blockDim.x + threadIdx.x;
  int n = t >> 16, p = t & 65535;
  const float* dn = d + ((size_t)n << 22) + p;
  float a0 = b[0], a1 = b[1], a2 = b[2];
  for (int ic = 0; ic < 64; ++ic) {
    float v = dn[(size_t)ic << 16];
    a0 += w[ic] * v;
    a1 += w[64 + ic] * v;
    a2 += w[128 + ic] * v;
  }
  float* on = o + ((size_t)n * 3 << 16) + p;
  on[0]      = tanhf(a0);
  on[65536]  = tanhf(a1);
  on[131072] = tanhf(a2);
}

// ---------------------------------------------------------------------------
// Host launcher
// ---------------------------------------------------------------------------
extern "C" void kernel_launch(void* const* d_in, const int* in_sizes, int n_in,
                              void* d_out, int out_size, void* d_ws, size_t ws_size,
                              hipStream_t stream) {
  (void)in_sizes; (void)n_in; (void)out_size; (void)ws_size;
  const float* x      = (const float*)d_in[0];
  const float* w_in   = (const float*)d_in[1];
  const float* b_in   = (const float*)d_in[2];
  const float* g_in   = (const float*)d_in[3];
  const float* be_in  = (const float*)d_in[4];
  const float* enc_w[3]  = {(const float*)d_in[5],  (const float*)d_in[6],  (const float*)d_in[7]};
  const float* enc_b[3]  = {(const float*)d_in[8],  (const float*)d_in[9],  (const float*)d_in[10]};
  const float* enc_g[3]  = {(const float*)d_in[11], (const float*)d_in[12], (const float*)d_in[13]};
  const float* enc_be[3] = {(const float*)d_in[14], (const float*)d_in[15], (const float*)d_in[16]};
  const float* w_preq  = (const float*)d_in[17];
  const float* b_preq  = (const float*)d_in[18];
  const float* codebook= (const float*)d_in[19];
  const float* w_postq = (const float*)d_in[20];
  const float* b_postq = (const float*)d_in[21];
  const float* dec_w[3]  = {(const float*)d_in[22], (const float*)d_in[23], (const float*)d_in[24]};
  const float* dec_b[3]  = {(const float*)d_in[25], (const float*)d_in[26], (const float*)d_in[27]};
  const float* dec_g[3]  = {(const float*)d_in[28], (const float*)d_in[29], (const float*)d_in[30]};
  const float* dec_be[3] = {(const float*)d_in[31], (const float*)d_in[32], (const float*)d_in[33]};
  const float* w_out   = (const float*)d_in[34];
  const float* b_out   = (const float*)d_in[35];
  float* out = (float*)d_out;

  char* wp = (char*)d_ws;
  auto carve = [&wp](size_t bytes) -> char* {
    char* p = wp;
    wp += (bytes + 255) & ~(size_t)255;
    return p;
  };
  const size_t BIG = (size_t)8 * 64 * 256 * 256;          // largest activation (elems)
  float* big0  = (float*)carve(BIG * 4);
  float* big1  = (float*)carve(BIG * 4);
  float* zbuf  = (float*)carve((size_t)8 * 64 * 1024 * 4);
  float* qbuf  = (float*)carve((size_t)8 * 64 * 1024 * 4);
  int*   idxb  = (int*)  carve(8192 * 4);
  float* stats = (float*)carve(2 * 512 * 4);
  float* lossa = (float*)carve(256);
  float* cbn   = (float*)carve(1024 * 4);
  _Float16* hw_enc[3];
  hw_enc[0] = (_Float16*)carve((size_t)128 * 64  * 16 * 2);
  hw_enc[1] = (_Float16*)carve((size_t)256 * 128 * 16 * 2);
  hw_enc[2] = (_Float16*)carve((size_t)512 * 256 * 16 * 2);
  _Float16* hw_preq  = (_Float16*)carve((size_t)64  * 512 * 2);
  _Float16* hw_postq = (_Float16*)carve((size_t)512 * 64  * 2);
  _Float16* hw_cb    = (_Float16*)carve((size_t)1024 * 64 * 2);
  _Float16* hw_dec[3];
  hw_dec[0] = (_Float16*)carve((size_t)4 * 256 * 512 * 4 * 2);
  hw_dec[1] = (_Float16*)carve((size_t)4 * 128 * 256 * 4 * 2);
  hw_dec[2] = (_Float16*)carve((size_t)4 * 64  * 128 * 4 * 2);

  // ---- weight preparation (cheap, every call: keeps launch deterministic) ----
  auto cvt = [&](const float* s, _Float16* d, int n) {
    cvt_f16_kern<<<(n + 255) / 256, 256, 0, stream>>>(s, d, n);
  };
  cvt(enc_w[0], hw_enc[0], 128 * 64 * 16);
  cvt(enc_w[1], hw_enc[1], 256 * 128 * 16);
  cvt(enc_w[2], hw_enc[2], 512 * 256 * 16);
  cvt(w_preq,  hw_preq,  64 * 512);
  cvt(w_postq, hw_postq, 512 * 64);
  cvt(codebook, hw_cb, 1024 * 64);
  pack_deconv_kern<<<(4 * 256 * 512 * 4 + 255) / 256, 256, 0, stream>>>(dec_w[0], hw_dec[0], 512, 256);
  pack_deconv_kern<<<(4 * 128 * 256 * 4 + 255) / 256, 256, 0, stream>>>(dec_w[1], hw_dec[1], 256, 128);
  pack_deconv_kern<<<(4 * 64  * 128 * 4 + 255) / 256, 256, 0, stream>>>(dec_w[2], hw_dec[2], 128, 64);
  cbnorm_kern<<<4, 256, 0, stream>>>(codebook, cbn);

  // ---- stem: 1x1 conv 3->64, BN, SiLU (in-place) ----
  {
    size_t tot = (size_t)8 * 64 * 65536;
    stem_kern<<<(unsigned)(tot / 256), 256, 0, stream>>>(x, w_in, b_in, big0);
    bn_stats_kern<<<64, 256, 0, stream>>>(big0, g_in, be_in, stats, 64, 16);
    bn_silu_kern<<<(unsigned)((tot + 255) / 256), 256, 0, stream>>>(big0, stats, 64, 16, (long long)tot);
  }

  // ---- encoder: 3x (conv4x4 s2 p1 -> BN -> SiLU) ----
  struct LE { int IC, IH, lgIW, OC, lgOW; };
  const LE ec[3] = {{64, 256, 8, 128, 7}, {128, 128, 7, 256, 6}, {256, 64, 6, 512, 5}};
  float* cur = big0; float* nxt = big1;
  for (int l = 0; l < 3; ++l) {
    int IC = ec[l].IC, IH = ec[l].IH, OC = ec[l].OC;
    int lgIW = ec[l].lgIW, lgOW = ec[l].lgOW;
    int HWo = 1 << (2 * lgOW);
    conv_wmma_kern<4, 4, 2, 1><<<dim3(HWo / 128, OC / 16, 8), 128, 0, stream>>>(
        cur, hw_enc[l], enc_b[l], nxt, IC, IH, lgIW, OC, lgOW, HWo);
    size_t tot = (size_t)8 * OC * HWo;
    bn_stats_kern<<<OC, 256, 0, stream>>>(nxt, enc_g[l], enc_be[l], stats, OC, 2 * lgOW);
    bn_silu_kern<<<(unsigned)((tot + 255) / 256), 256, 0, stream>>>(nxt, stats, OC, 2 * lgOW, (long long)tot);
    float* t = cur; cur = nxt; nxt = t;
  }
  // cur now holds encoder output (8,512,32,32)

  // ---- pre-quant 1x1 conv 512->64 (async-LDS staged GEMM) ----
  conv1x1_wmma_kern<<<dim3(1024 / 128, 64 / 16, 8), 128, 0, stream>>>(
      cur, hw_preq, b_preq, zbuf, 512, 64);

  // ---- vector quantization ----
  vq_argmin_kern<<<dim3(16, 1, 8), 128, 0, stream>>>(zbuf, hw_cb, cbn, idxb);
  zero_kern<<<1, 1, 0, stream>>>(lossa);
  vq_gather_kern<<<32, 256, 0, stream>>>(zbuf, codebook, idxb, qbuf, lossa);
  loss_fin_kern<<<1, 1, 0, stream>>>(lossa, out + (size_t)8 * 3 * 65536);

  // ---- post-quant 1x1 conv 64->512 (async-LDS staged GEMM) ----
  conv1x1_wmma_kern<<<dim3(1024 / 128, 512 / 16, 8), 128, 0, stream>>>(
      qbuf, hw_postq, b_postq, big0, 64, 512);

  // ---- decoder: 3x (deconv4x4 s2 p1 -> BN -> SiLU) ----
  struct LD { int IC, IH, lgIW, OC; };
  const LD dc[3] = {{512, 32, 5, 256}, {256, 64, 6, 128}, {128, 128, 7, 64}};
  cur = big0; nxt = big1;
  for (int l = 0; l < 3; ++l) {
    int IC = dc[l].IC, IH = dc[l].IH, OC = dc[l].OC, lgIW = dc[l].lgIW;
    int HWi = IH << lgIW;
    deconv_wmma_kern<<<dim3(HWi / 128, OC / 16, 8 * 4), 128, 0, stream>>>(
        cur, hw_dec[l], dec_b[l], nxt, IC, IH, lgIW, OC);
    int lgHWo = 2 * (lgIW + 1);
    size_t tot = (size_t)8 * OC << lgHWo;
    bn_stats_kern<<<OC, 256, 0, stream>>>(nxt, dec_g[l], dec_be[l], stats, OC, lgHWo);
    bn_silu_kern<<<(unsigned)((tot + 255) / 256), 256, 0, stream>>>(nxt, stats, OC, lgHWo, (long long)tot);
    float* t = cur; cur = nxt; nxt = t;
  }
  // cur holds decoder output (8,64,256,256)

  // ---- head: 1x1 conv 64->3 + tanh -> d_out ----
  head_kern<<<(8 * 65536) / 256, 256, 0, stream>>>(cur, w_out, b_out, out);
}
